// quality_scoring_4827543241294
// MI455X (gfx1250) — matrix operand order repaired and checked by reading, not compiled, not run on hardware
//
#include <hip/hip_runtime.h>
#include <math.h>

// ---- problem constants (from reference) ----
#define EMBED   768
#define HID     384
#define NTOK    (32 * 1024)      // B*N tokens
#define MTILE   64               // tokens per workgroup (2 waves x 2 strips x 16)
#define KSTEPS  (EMBED / 32)     // 24 k-chunks of 32
#define NSTEPS  (HID / 16)       // 24 n-tiles of 16
#define WELEMS  (EMBED * HID)    // 294912 per weight matrix
#define STRIPSTRIDE (KSTEPS * 512)   // bf16 elems between 16-token strips in LDS

typedef __attribute__((ext_vector_type(16))) __bf16 v16bf;
typedef __attribute__((ext_vector_type(8)))  float  v8f;
typedef __attribute__((ext_vector_type(4)))  float  v4f;

struct alignas(8) US4 { unsigned short x, y, z, w; };

// round-to-nearest-even f32 -> bf16 (bit pattern)
static __device__ __forceinline__ unsigned short f2bf(float f) {
    unsigned int u = __float_as_uint(f);
    u += 0x7FFFu + ((u >> 16) & 1u);
    return (unsigned short)(u >> 16);
}

// -------------------------------------------------------------------------
// Pack Ws1/Ww1 (fp32 [E,H] row-major) into bf16 WMMA B-fragment order.
// Fragment (t = n0/16, c = k0/32): 32 lanes x 16 bf16 contiguous.
// B layout (v_wmma_*_16x16x32_bf16): lane = (n&15) + 16*((kk>>3)&1), elem j = (kk&7) + 8*(kk>>4).
// Also zero the 64 per-batch accumulators at the front of d_ws.
// -------------------------------------------------------------------------
__global__ void qs_pack_kernel(const float* __restrict__ Ws1,
                               const float* __restrict__ Ww1,
                               unsigned short* __restrict__ wS,
                               unsigned short* __restrict__ wW,
                               float* __restrict__ acc) {
    int idx = blockIdx.x * 256 + threadIdx.x;
    if (idx < 64) acc[idx] = 0.0f;
    if (idx >= WELEMS) return;
    int e = idx / HID, h = idx - e * HID;
    int c = e >> 5, kk = e & 31;
    int half = (kk >> 3) & 1, j = (kk & 7) + ((kk >> 4) << 3);
    int t = h >> 4, lane = (h & 15) + (half << 4);
    int dst = ((t * KSTEPS + c) * 32 + lane) * 16 + j;
    wS[dst] = f2bf(Ws1[idx]);
    wW[dst] = f2bf(Ww1[idx]);
}

// -------------------------------------------------------------------------
// Main fused kernel: 512 WGs x 64 threads (2 waves). Each wave owns TWO
// 16-token strips (MS=2) and processes TWO n-tiles per k-sweep (NT=2):
// per k-step, 2 A-fragment ds_loads + 4 B-fragment global loads feed
// 8 independent WMMA chains (2 strips x 2 n-tiles x 2 branches).
// x tile staged in LDS in A-fragment order (16-bit A layout:
// lane = (row&15)+16*hi, hi=(kk>>3)&1, elem j=(kk&7)+8*((kk>>4)&1)).
// -------------------------------------------------------------------------
__global__ void qs_main_kernel(const float* __restrict__ x,
                               const unsigned short* __restrict__ wS,
                               const unsigned short* __restrict__ wW,
                               const float* __restrict__ bs1,
                               const float* __restrict__ Ws2,
                               const float* __restrict__ bs2,
                               const float* __restrict__ bw1,
                               const float* __restrict__ Ww2,
                               const float* __restrict__ bw2,
                               float* __restrict__ accFW,
                               float* __restrict__ accW) {
    extern __shared__ unsigned short ldsx[];   // 64 tokens * 768 * bf16 = 96 KB
    const int tid  = threadIdx.x;
    const int wg   = blockIdx.x;
    const int lane = tid & 31;
    const int w    = tid >> 5;                 // wave id 0..1 -> strips 2w, 2w+1

    // ---- stage + convert + swizzle x tile into LDS (A-fragment order) ----
    // x is read exactly once device-wide -> non-temporal (keep L2 for weights)
    const int NF4 = MTILE * EMBED / 4;         // 12288 float4
    for (int i = tid; i < NF4; i += 64) {
        int m = i / (EMBED / 4);
        int q = i - m * (EMBED / 4);
        int k = q * 4;
        v4f v = __builtin_nontemporal_load(
                    reinterpret_cast<const v4f*>(
                        x + (size_t)(wg * MTILE + m) * EMBED) + q);
        int ss = m >> 4, r = m & 15;
        int c = k >> 5, kk = k & 31;
        int hi = (kk >> 3) & 1;
        int j  = (kk & 7) + (((kk >> 4) & 1) << 3);
        int ln = r + (hi << 4);
        int off = ((ss * KSTEPS + c) * 32 + ln) * 16 + j;
        US4 p; p.x = f2bf(v.x); p.y = f2bf(v.y); p.z = f2bf(v.z); p.w = f2bf(v.w);
        *reinterpret_cast<US4*>(&ldsx[off]) = p;
    }
    __syncthreads();

    // ---- GEMM loops: per-token scalar accumulators, never materialize hs ----
    float pS[2][8], pW[2][8];
#pragma unroll
    for (int u = 0; u < 2; ++u)
#pragma unroll
        for (int m = 0; m < 8; ++m) { pS[u][m] = 0.0f; pW[u][m] = 0.0f; }

    // strip pair for this wave: s0 = 2w, s1 = 2w+1 (24 KB apart in LDS)
    const unsigned short* aBase = ldsx + (((2 * w) * KSTEPS) * 32 + lane) * 16;
    const int fragStride = KSTEPS * 512;       // bf16 elems between n-tiles of W

#pragma unroll 1
    for (int tp = 0; tp < NSTEPS / 2; ++tp) {
        const int t0 = tp * 2;
        v8f cs00 = {}, cs01 = {}, cs10 = {}, cs11 = {};
        v8f cw00 = {}, cw01 = {}, cw10 = {}, cw11 = {};
        const unsigned short* aP  = aBase;
        const unsigned short* bS0 = wS + (size_t)t0 * fragStride + lane * 16;
        const unsigned short* bW0 = wW + (size_t)t0 * fragStride + lane * 16;
#pragma unroll 2
        for (int c = 0; c < KSTEPS; ++c) {
            v16bf A0  = *reinterpret_cast<const v16bf*>(aP);
            v16bf A1  = *reinterpret_cast<const v16bf*>(aP + STRIPSTRIDE);
            v16bf Bs0 = *reinterpret_cast<const v16bf*>(bS0);
            v16bf Bs1 = *reinterpret_cast<const v16bf*>(bS0 + fragStride);
            v16bf Bw0 = *reinterpret_cast<const v16bf*>(bW0);
            v16bf Bw1 = *reinterpret_cast<const v16bf*>(bW0 + fragStride);
            cs00 = __builtin_amdgcn_wmma_f32_16x16x32_bf16(
                       false, A0, false, Bs0, (short)0, cs00, false, false);
            cs10 = __builtin_amdgcn_wmma_f32_16x16x32_bf16(
                       false, A1, false, Bs0, (short)0, cs10, false, false);
            cw00 = __builtin_amdgcn_wmma_f32_16x16x32_bf16(
                       false, A0, false, Bw0, (short)0, cw00, false, false);
            cw10 = __builtin_amdgcn_wmma_f32_16x16x32_bf16(
                       false, A1, false, Bw0, (short)0, cw10, false, false);
            cs01 = __builtin_amdgcn_wmma_f32_16x16x32_bf16(
                       false, A0, false, Bs1, (short)0, cs01, false, false);
            cs11 = __builtin_amdgcn_wmma_f32_16x16x32_bf16(
                       false, A1, false, Bs1, (short)0, cs11, false, false);
            cw01 = __builtin_amdgcn_wmma_f32_16x16x32_bf16(
                       false, A0, false, Bw1, (short)0, cw01, false, false);
            cw11 = __builtin_amdgcn_wmma_f32_16x16x32_bf16(
                       false, A1, false, Bw1, (short)0, cw11, false, false);
            aP += 512; bS0 += 512; bW0 += 512;
        }
        // C layout: lane holds column n = t*16 + (lane&15); VGPR m = row m
        // (rows 8..15 in lanes 16..31). Fold both n-tiles into pS/pW per strip.
        int n0 = t0 * 16 + (lane & 15);
        int n1 = n0 + 16;
        float b1s0 = bs1[n0], w2s0 = Ws2[n0], b1w0 = bw1[n0], w2w0 = Ww2[n0];
        float b1s1 = bs1[n1], w2s1 = Ws2[n1], b1w1 = bw1[n1], w2w1 = Ww2[n1];
#pragma unroll
        for (int m = 0; m < 8; ++m) {
            float h;
            h = cs00[m] + b1s0; h = h > 0.0f ? h : 0.0f; pS[0][m] += h * w2s0;
            h = cs01[m] + b1s1; h = h > 0.0f ? h : 0.0f; pS[0][m] += h * w2s1;
            h = cs10[m] + b1s0; h = h > 0.0f ? h : 0.0f; pS[1][m] += h * w2s0;
            h = cs11[m] + b1s1; h = h > 0.0f ? h : 0.0f; pS[1][m] += h * w2s1;
            h = cw00[m] + b1w0; h = h > 0.0f ? h : 0.0f; pW[0][m] += h * w2w0;
            h = cw01[m] + b1w1; h = h > 0.0f ? h : 0.0f; pW[0][m] += h * w2w1;
            h = cw10[m] + b1w0; h = h > 0.0f ? h : 0.0f; pW[1][m] += h * w2w0;
            h = cw11[m] + b1w1; h = h > 0.0f ? h : 0.0f; pW[1][m] += h * w2w1;
        }
    }

    // ---- reduce the 16 columns held by each 16-lane half ----
#pragma unroll
    for (int u = 0; u < 2; ++u)
#pragma unroll
        for (int m = 0; m < 8; ++m) {
#pragma unroll
            for (int d = 1; d < 16; d <<= 1) {
                pS[u][m] += __shfl_xor(pS[u][m], d, 32);
                pW[u][m] += __shfl_xor(pW[u][m], d, 32);
            }
        }

    // lane 0 holds rows 0..7 of each strip, lane 16 holds rows 8..15
    if (lane == 0 || lane == 16) {
        float bs2v = bs2[0], bw2v = bw2[0];
#pragma unroll
        for (int u = 0; u < 2; ++u) {
            float sfw = 0.0f, sw = 0.0f;
#pragma unroll
            for (int m = 0; m < 8; ++m) {
                float f = pS[u][m] + bs2v; f = f > 0.0f ? f : 0.0f;
                float ww = 1.0f / (1.0f + __expf(-(pW[u][m] + bw2v)));
                sfw += f * ww;
                sw  += ww;
            }
            int token0 = wg * MTILE + (2 * w + u) * 16 + ((lane >> 4) << 3);
            int b = token0 >> 10;             // N = 1024 tokens per batch item
            atomicAdd(&accFW[b], sfw);
            atomicAdd(&accW[b],  sw);
        }
    }
}

__global__ void qs_finalize_kernel(const float* __restrict__ acc,
                                   float* __restrict__ out) {
    int b = threadIdx.x;
    if (b < 32) out[b] = acc[b] / acc[32 + b];
}

extern "C" void kernel_launch(void* const* d_in, const int* in_sizes, int n_in,
                              void* d_out, int out_size, void* d_ws, size_t ws_size,
                              hipStream_t stream) {
    const float* x   = (const float*)d_in[0];
    const float* Ws1 = (const float*)d_in[1];
    const float* bs1 = (const float*)d_in[2];
    const float* Ws2 = (const float*)d_in[3];
    const float* bs2 = (const float*)d_in[4];
    const float* Ww1 = (const float*)d_in[5];
    const float* bw1 = (const float*)d_in[6];
    const float* Ww2 = (const float*)d_in[7];
    const float* bw2 = (const float*)d_in[8];

    // d_ws layout: [0,256) per-batch accumulators (sumFW[32], sumW[32]);
    // then bf16 pre-swizzled Ws1 and Ww1 (294912 elems each).
    float* acc = (float*)d_ws;
    unsigned short* wS = (unsigned short*)((char*)d_ws + 256);
    unsigned short* wW = wS + WELEMS;

    // allow 96 KB dynamic LDS (320 KB/WGP on CDNA5; default cap is lower)
    hipFuncSetAttribute((const void*)qs_main_kernel,
                        hipFuncAttributeMaxDynamicSharedMemorySize,
                        MTILE * EMBED * 2);

    qs_pack_kernel<<<(WELEMS + 255) / 256, 256, 0, stream>>>(Ws1, Ww1, wS, wW, acc);
    qs_main_kernel<<<NTOK / MTILE, 64, MTILE * EMBED * 2, stream>>>(
        x, wS, wW, bs1, Ws2, bs2, bw1, Ww2, bw2, acc, acc + 32);
    qs_finalize_kernel<<<1, 32, 0, stream>>>(acc, (float*)d_out);
}
